// MambaEncoderLayer_8409545966072
// MI455X (gfx1250) — compile-verified, matrix operand order checked by reading, not run on hardware
//
#include <hip/hip_runtime.h>
#include <hip/hip_bf16.h>
#include <math.h>

// ---------------------------------------------------------------------------
// Types
// ---------------------------------------------------------------------------
typedef __bf16 v16bf __attribute__((ext_vector_type(16)));
typedef __bf16 v8bf  __attribute__((ext_vector_type(8)));
typedef float  v8f   __attribute__((ext_vector_type(8)));

#define D_MODEL 1024
#define FFN_DIM 4096
#define D_STATE 16
#define D_CONV  4
#define D_INNER 2048
#define DT_RANK 64
#define BB 2
#define LL 1024
#define ROWS (BB * LL)   // 2048

// ---------------------------------------------------------------------------
// Small helpers
// ---------------------------------------------------------------------------
__device__ __forceinline__ float silu_f(float x) { return x / (1.0f + __expf(-x)); }
__device__ __forceinline__ float softplus_f(float x) {
    return (x > 20.0f) ? x : log1pf(__expf(x));
}
__device__ __forceinline__ float gelu_tanh_f(float x) {
    const float k = 0.7978845608028654f; // sqrt(2/pi)
    float x3 = x * x * x;
    return 0.5f * x * (1.0f + tanhf(k * (x + 0.044715f * x3)));
}

__device__ __forceinline__ v16bf ld_frag(const __bf16* p) {
    v8bf lo = *(const v8bf*)p;
    v8bf hi = *(const v8bf*)(p + 8);
    return __builtin_shufflevector(lo, hi, 0, 1, 2, 3, 4, 5, 6, 7,
                                           8, 9, 10, 11, 12, 13, 14, 15);
}

// CDNA5 async global->LDS copy (ASYNCcnt-tracked, bypasses VGPRs).
// lds_addr: byte offset within the workgroup's LDS allocation (low 32 bits of
// the generic pointer); gaddr: 64-bit global address. GV addressing mode.
__device__ __forceinline__ void async_copy_b128(unsigned lds_addr,
                                                const __bf16* g) {
    asm volatile("global_load_async_to_lds_b128 %0, %1, off"
                 :: "v"(lds_addr), "v"((unsigned long long)(size_t)g)
                 : "memory");
}
__device__ __forceinline__ void wait_async0() {
    asm volatile("s_wait_asynccnt 0" ::: "memory");
}

// ---------------------------------------------------------------------------
// f32 -> bf16 conversion (weights + activations)
// ---------------------------------------------------------------------------
__global__ void f2bf_kernel(const float* __restrict__ in, __bf16* __restrict__ out,
                            unsigned int n) {
    unsigned int i = blockIdx.x * 256u + threadIdx.x;
    if (i < n) out[i] = (__bf16)in[i];
}

// ---------------------------------------------------------------------------
// Fused LN0 + LN1 : x -> x0 (f32, residual) and u = LN1(x0) (bf16)
// ---------------------------------------------------------------------------
__global__ __launch_bounds__(256)
void ln0_ln1_kernel(const float* __restrict__ x,
                    const float* __restrict__ w0, const float* __restrict__ b0,
                    const float* __restrict__ w1, const float* __restrict__ b1,
                    float* __restrict__ x0, __bf16* __restrict__ u) {
    const int C = D_MODEL;
    int row = blockIdx.x;
    int t = threadIdx.x;
    const float* xr = x + (size_t)row * C;

    __shared__ float rs[256], rq[256];

    float v[4];
    float s = 0.f, sq = 0.f;
#pragma unroll
    for (int j = 0; j < 4; ++j) {
        v[j] = xr[t + j * 256];
        s += v[j];
        sq += v[j] * v[j];
    }
    rs[t] = s; rq[t] = sq;
    __syncthreads();
    for (int o = 128; o > 0; o >>= 1) {
        if (t < o) { rs[t] += rs[t + o]; rq[t] += rq[t + o]; }
        __syncthreads();
    }
    float mu = rs[0] * (1.0f / C);
    float var = rq[0] * (1.0f / C) - mu * mu;
    float rstd = rsqrtf(var + 1e-5f);
    __syncthreads();

    float y[4];
    s = 0.f; sq = 0.f;
#pragma unroll
    for (int j = 0; j < 4; ++j) {
        int c = t + j * 256;
        y[j] = (v[j] - mu) * rstd * w0[c] + b0[c];
        x0[(size_t)row * C + c] = y[j];
        s += y[j];
        sq += y[j] * y[j];
    }
    rs[t] = s; rq[t] = sq;
    __syncthreads();
    for (int o = 128; o > 0; o >>= 1) {
        if (t < o) { rs[t] += rs[t + o]; rq[t] += rq[t + o]; }
        __syncthreads();
    }
    float mu2 = rs[0] * (1.0f / C);
    float var2 = rq[0] * (1.0f / C) - mu2 * mu2;
    float rstd2 = rsqrtf(var2 + 1e-5f);
#pragma unroll
    for (int j = 0; j < 4; ++j) {
        int c = t + j * 256;
        u[(size_t)row * C + c] = (__bf16)((y[j] - mu2) * rstd2 * w1[c] + b1[c]);
    }
}

// ---------------------------------------------------------------------------
// Single LN (LN2): x1 -> h (bf16)
// ---------------------------------------------------------------------------
__global__ __launch_bounds__(256)
void ln_single_kernel(const float* __restrict__ x,
                      const float* __restrict__ w, const float* __restrict__ b,
                      __bf16* __restrict__ out) {
    const int C = D_MODEL;
    int row = blockIdx.x;
    int t = threadIdx.x;
    const float* xr = x + (size_t)row * C;
    __shared__ float rs[256], rq[256];
    float v[4];
    float s = 0.f, sq = 0.f;
#pragma unroll
    for (int j = 0; j < 4; ++j) {
        v[j] = xr[t + j * 256];
        s += v[j];
        sq += v[j] * v[j];
    }
    rs[t] = s; rq[t] = sq;
    __syncthreads();
    for (int o = 128; o > 0; o >>= 1) {
        if (t < o) { rs[t] += rs[t + o]; rq[t] += rq[t + o]; }
        __syncthreads();
    }
    float mu = rs[0] * (1.0f / C);
    float var = rq[0] * (1.0f / C) - mu * mu;
    float rstd = rsqrtf(var + 1e-5f);
#pragma unroll
    for (int j = 0; j < 4; ++j) {
        int c = t + j * 256;
        out[(size_t)row * C + c] = (__bf16)((v[j] - mu) * rstd * w[c] + b[c]);
    }
}

// ---------------------------------------------------------------------------
// Tiled WMMA GEMM:  C[M,N] = epilogue( A[M,K] * W[N,K]^T + bias + res )
//   bf16 inputs, f32 accumulate via v_wmma_f32_16x16x32_bf16.
//   mode: 0 = linear, 1 = softplus, 2 = tanh-GELU; Cf/Cb optional outputs.
// Block: 256 threads = 8 waves; block tile 64x128; wave tile 32x32 (2x2 WMMA).
// Double-buffered LDS fed by global_load_async_to_lds_b128 (ASYNCcnt),
// BK=64 -> 8 WMMAs per K-iteration, one barrier per iteration.
// ---------------------------------------------------------------------------
#define BM 64
#define BN 128
#define BK 64
#define LDP 72   // padded LDS row stride (bf16 elems): 36 banks/row

__global__ __launch_bounds__(256)
void gemm_wmma_kernel(const __bf16* __restrict__ A, const __bf16* __restrict__ W,
                      const float* __restrict__ bias, const float* __restrict__ res,
                      float* __restrict__ Cf, __bf16* __restrict__ Cb,
                      int M, int N, int K, int mode) {
    __shared__ __bf16 sA[2][BM][LDP];   // 18 KB
    __shared__ __bf16 sW[2][BN][LDP];   // 36 KB

    int t = threadIdx.x;
    int lane = t & 31;
    int wv = t >> 5;
    int warpM = wv >> 2;        // 0..1
    int warpN = wv & 3;         // 0..3
    int bM = blockIdx.y * BM;
    int bN = blockIdx.x * BN;

    v8f acc[2][2];
#pragma unroll
    for (int i = 0; i < 2; ++i)
#pragma unroll
        for (int j = 0; j < 2; ++j)
#pragma unroll
            for (int r = 0; r < 8; ++r) acc[i][j][r] = 0.0f;

    // Pre-zero out-of-range W rows (they are never overwritten: the async
    // copies for those lanes are EXEC-masked off in every stage).
    if (bN + BN > N) {
        v8bf z;
#pragma unroll
        for (int q = 0; q < 8; ++q) z[q] = (__bf16)0.0f;
#pragma unroll
        for (int s = 0; s < 4; ++s) {
            int slot = t + s * 256;
            int row = slot >> 3;
            int col = (slot & 7) * 8;
            if (bN + row >= N) {
                *(v8bf*)&sW[0][row][col] = z;
                *(v8bf*)&sW[1][row][col] = z;
            }
        }
        __syncthreads();
    }

    // Issue one stage of async global->LDS copies (6 instructions / wave).
    auto stage = [&](int buf, int k0) {
#pragma unroll
        for (int s = 0; s < 2; ++s) {           // A tile: 64x64 bf16 = 8 KB
            int slot = t + s * 256;
            int row = slot >> 3;
            int col = (slot & 7) * 8;
            async_copy_b128((unsigned)(size_t)&sA[buf][row][col],
                            A + (size_t)(bM + row) * K + k0 + col);
        }
#pragma unroll
        for (int s = 0; s < 4; ++s) {           // W tile: 128x64 bf16 = 16 KB
            int slot = t + s * 256;
            int row = slot >> 3;
            int col = (slot & 7) * 8;
            int gn = bN + row;
            if (gn < N)
                async_copy_b128((unsigned)(size_t)&sW[buf][row][col],
                                W + (size_t)gn * K + k0 + col);
        }
    };

    stage(0, 0);

    int lsel = lane & 15;
    int kh = (lane >> 4) * 16;  // 0 or 16

    int buf = 0;
    for (int k0 = 0; k0 < K; k0 += BK, buf ^= 1) {
        wait_async0();          // this wave's copies into sA/sW[buf] landed
        __syncthreads();        // => all waves' copies landed; prev reads done
        if (k0 + BK < K)
            stage(buf ^ 1, k0 + BK);   // overlap next loads with this math
        if (k0 + 2 * BK < K) {         // deep prefetch into L2
            __builtin_prefetch(A + (size_t)(bM + (t >> 3)) * K + k0 + 2 * BK, 0, 1);
            __builtin_prefetch(W + (size_t)min(bN + (t >> 3), N - 1) * K + k0 + 2 * BK, 0, 1);
        }

#pragma unroll
        for (int kk = 0; kk < 2; ++kk) {       // two 32-deep K slabs
            v16bf aF[2], bF[2];
#pragma unroll
            for (int i = 0; i < 2; ++i)
                aF[i] = ld_frag(&sA[buf][warpM * 32 + i * 16 + lsel][kk * 32 + kh]);
#pragma unroll
            for (int j = 0; j < 2; ++j)
                bF[j] = ld_frag(&sW[buf][warpN * 32 + j * 16 + lsel][kk * 32 + kh]);
#pragma unroll
            for (int i = 0; i < 2; ++i)
#pragma unroll
                for (int j = 0; j < 2; ++j)
                    acc[i][j] = __builtin_amdgcn_wmma_f32_16x16x32_bf16(
                        false, aF[i], false, bF[j], (short)0, acc[i][j],
                        false, false);
        }
    }

    // Epilogue: VGPR r -> (M = r + 8*(lane>>4), N = lane&15) within 16x16 tile
    int mo = (lane >> 4) << 3;
    int no = lane & 15;
#pragma unroll
    for (int i = 0; i < 2; ++i) {
#pragma unroll
        for (int j = 0; j < 2; ++j) {
#pragma unroll
            for (int r = 0; r < 8; ++r) {
                int gm = bM + warpM * 32 + i * 16 + r + mo;
                int gn = bN + warpN * 32 + j * 16 + no;
                if (gm < M && gn < N) {
                    float v = acc[i][j][r];
                    if (bias) v += bias[gn];
                    if (res)  v += res[(size_t)gm * N + gn];
                    if (mode == 1)      v = softplus_f(v);
                    else if (mode == 2) v = gelu_tanh_f(v);
                    if (Cf) Cf[(size_t)gm * N + gn] = v;
                    if (Cb) Cb[(size_t)gm * N + gn] = (__bf16)v;
                }
            }
        }
    }
}

// ---------------------------------------------------------------------------
// Causal depthwise conv (width 4) + bias + SiLU.  xz is [ROWS, 2*D_INNER];
// xc occupies cols [0, D_INNER).  Writes f32 and bf16 copies.
// ---------------------------------------------------------------------------
__global__ void conv_silu_kernel(const float* __restrict__ xz,
                                 const float* __restrict__ cw,
                                 const float* __restrict__ cb,
                                 float* __restrict__ xcf, __bf16* __restrict__ xcb) {
    size_t idx = (size_t)blockIdx.x * 256 + threadIdx.x; // ROWS*D_INNER
    int d = (int)(idx & (D_INNER - 1));
    size_t rl = idx >> 11;
    int l = (int)(rl & (LL - 1));
    size_t b = rl >> 10;
    float acc = cb[d];
#pragma unroll
    for (int j = 0; j < D_CONV; ++j) {
        int li = l - (D_CONV - 1) + j;
        if (li >= 0)
            acc += cw[d * D_CONV + j] * xz[((b << 10) + li) * (2 * D_INNER) + d];
    }
    float s = silu_f(acc);
    xcf[idx] = s;
    xcb[idx] = (__bf16)s;
}

// ---------------------------------------------------------------------------
// Extract dt columns (first DT_RANK of x_dbl) as bf16
// ---------------------------------------------------------------------------
__global__ void extract_dt_kernel(const float* __restrict__ xdbl,
                                  __bf16* __restrict__ dt) {
    size_t idx = (size_t)blockIdx.x * 256 + threadIdx.x; // ROWS*DT_RANK
    int c = (int)(idx & (DT_RANK - 1));
    size_t row = idx >> 6;
    dt[idx] = (__bf16)xdbl[row * (DT_RANK + 2 * D_STATE) + c];
}

// ---------------------------------------------------------------------------
// Selective scan. Thread per (b, d); 16-element state in registers; B/C
// vectors broadcast through LDS per timestep. y overwrites delta in place.
// ---------------------------------------------------------------------------
__global__ __launch_bounds__(256)
void scan_kernel(const float* __restrict__ xdbl, const float* __restrict__ A_log,
                 const float* __restrict__ xc, float* __restrict__ delta_y) {
    int b = blockIdx.x >> 3;
    int d = ((blockIdx.x & 7) << 8) + threadIdx.x;

    float Ac[D_STATE], h[D_STATE];
#pragma unroll
    for (int n = 0; n < D_STATE; ++n) {
        Ac[n] = -__expf(A_log[d * D_STATE + n]);
        h[n] = 0.0f;
    }

    __shared__ float sB[D_STATE], sC[D_STATE];

    for (int l = 0; l < LL; ++l) {
        size_t row = (size_t)b * LL + l;
        __syncthreads();
        if (threadIdx.x < 2 * D_STATE) {
            float val = xdbl[row * (DT_RANK + 2 * D_STATE) + DT_RANK + threadIdx.x];
            if (threadIdx.x < D_STATE) sB[threadIdx.x] = val;
            else                       sC[threadIdx.x - D_STATE] = val;
        }
        __syncthreads();
        size_t off = row * D_INNER + d;
        float dl = delta_y[off];
        float xv = xc[off];
        float y = 0.0f;
#pragma unroll
        for (int n = 0; n < D_STATE; ++n) {
            float dA = __expf(dl * Ac[n]);
            h[n] = dA * h[n] + (dl * sB[n]) * xv;
            y += h[n] * sC[n];
        }
        delta_y[off] = y;
    }
}

// ---------------------------------------------------------------------------
// Gating: y = (y_scan + xc * D) * silu(z); write bf16 for out_proj GEMM
// ---------------------------------------------------------------------------
__global__ void gate_kernel(const float* __restrict__ yscan,
                            const float* __restrict__ xc,
                            const float* __restrict__ xz,
                            const float* __restrict__ Dskip,
                            __bf16* __restrict__ ybf) {
    size_t idx = (size_t)blockIdx.x * 256 + threadIdx.x; // ROWS*D_INNER
    int d = (int)(idx & (D_INNER - 1));
    size_t row = idx >> 11;
    float yy = yscan[idx] + xc[idx] * Dskip[d];
    float z = xz[row * (2 * D_INNER) + D_INNER + d];
    ybf[idx] = (__bf16)(yy * silu_f(z));
}

// ---------------------------------------------------------------------------
// Host launcher
// ---------------------------------------------------------------------------
static inline size_t align_up(size_t x) { return (x + 255) & ~(size_t)255; }

extern "C" void kernel_launch(void* const* d_in, const int* in_sizes, int n_in,
                              void* d_out, int out_size, void* d_ws, size_t ws_size,
                              hipStream_t stream) {
    (void)in_sizes; (void)n_in; (void)out_size; (void)ws_size;

    const float* x        = (const float*)d_in[0];
    const float* ln0_w    = (const float*)d_in[1];
    const float* ln0_b    = (const float*)d_in[2];
    const float* ln1_w    = (const float*)d_in[3];
    const float* ln1_b    = (const float*)d_in[4];
    const float* ln2_w    = (const float*)d_in[5];
    const float* ln2_b    = (const float*)d_in[6];
    const float* in_proj_w  = (const float*)d_in[7];
    const float* conv_w     = (const float*)d_in[8];
    const float* conv_b     = (const float*)d_in[9];
    const float* x_proj_w   = (const float*)d_in[10];
    const float* dt_proj_w  = (const float*)d_in[11];
    const float* dt_proj_b  = (const float*)d_in[12];
    const float* A_log      = (const float*)d_in[13];
    const float* D_skip     = (const float*)d_in[14];
    const float* out_proj_w = (const float*)d_in[15];
    const float* ffn_w1     = (const float*)d_in[16];
    const float* ffn_b1     = (const float*)d_in[17];
    const float* ffn_w2     = (const float*)d_in[18];
    const float* ffn_b2     = (const float*)d_in[19];

    // ---- workspace carve-up ----
    char* ws = (char*)d_ws;
    size_t off = 0;
    auto carve = [&](size_t bytes) { char* p = ws + off; off += align_up(bytes); return p; };

    __bf16* w_in_bf   = (__bf16*)carve((size_t)2 * D_INNER * D_MODEL * 2);
    __bf16* w_xp_bf   = (__bf16*)carve((size_t)(DT_RANK + 2 * D_STATE) * D_INNER * 2);
    __bf16* w_dt_bf   = (__bf16*)carve((size_t)D_INNER * DT_RANK * 2);
    __bf16* w_op_bf   = (__bf16*)carve((size_t)D_MODEL * D_INNER * 2);
    __bf16* w_f1_bf   = (__bf16*)carve((size_t)FFN_DIM * D_MODEL * 2);
    __bf16* w_f2_bf   = (__bf16*)carve((size_t)D_MODEL * FFN_DIM * 2);
    float*  x0_f      = (float*) carve((size_t)ROWS * D_MODEL * 4);
    __bf16* u_bf      = (__bf16*)carve((size_t)ROWS * D_MODEL * 2);
    float*  xz_f      = (float*) carve((size_t)ROWS * 2 * D_INNER * 4);
    float*  xc_f      = (float*) carve((size_t)ROWS * D_INNER * 4);
    __bf16* xc_bf     = (__bf16*)carve((size_t)ROWS * D_INNER * 2);
    float*  xdbl_f    = (float*) carve((size_t)ROWS * (DT_RANK + 2 * D_STATE) * 4);
    __bf16* dt_bf     = (__bf16*)carve((size_t)ROWS * DT_RANK * 2);
    float*  delta_f   = (float*) carve((size_t)ROWS * D_INNER * 4);   // delta -> y
    __bf16* y_bf      = (__bf16*)carve((size_t)ROWS * D_INNER * 2);
    float*  x1_f      = (float*) carve((size_t)ROWS * D_MODEL * 4);
    __bf16* h_bf      = (__bf16*)carve((size_t)ROWS * D_MODEL * 2);
    __bf16* ffnh_bf   = (__bf16*)carve((size_t)ROWS * FFN_DIM * 2);

    // ---- weight conversion to bf16 ----
    auto conv_launch = [&](const float* src, __bf16* dst, unsigned int n) {
        f2bf_kernel<<<(n + 255) / 256, 256, 0, stream>>>(src, dst, n);
    };
    conv_launch(in_proj_w,  w_in_bf, 2u * D_INNER * D_MODEL);
    conv_launch(x_proj_w,   w_xp_bf, (DT_RANK + 2 * D_STATE) * D_INNER);
    conv_launch(dt_proj_w,  w_dt_bf, D_INNER * DT_RANK);
    conv_launch(out_proj_w, w_op_bf, D_MODEL * D_INNER);
    conv_launch(ffn_w1,     w_f1_bf, FFN_DIM * D_MODEL);
    conv_launch(ffn_w2,     w_f2_bf, D_MODEL * FFN_DIM);

    // ---- LN0 + LN1 ----
    ln0_ln1_kernel<<<ROWS, 256, 0, stream>>>(x, ln0_w, ln0_b, ln1_w, ln1_b, x0_f, u_bf);

    // ---- in_proj: xz[2048,4096] = u[2048,1024] * Win[4096,1024]^T ----
    gemm_wmma_kernel<<<dim3((2 * D_INNER) / BN, ROWS / BM), 256, 0, stream>>>(
        u_bf, w_in_bf, nullptr, nullptr, xz_f, nullptr,
        ROWS, 2 * D_INNER, D_MODEL, 0);

    // ---- depthwise conv + silu ----
    conv_silu_kernel<<<(ROWS * D_INNER) / 256, 256, 0, stream>>>(
        xz_f, conv_w, conv_b, xc_f, xc_bf);

    // ---- x_proj: x_dbl[2048,96] = xc[2048,2048] * Wxp[96,2048]^T ----
    gemm_wmma_kernel<<<dim3(1, ROWS / BM), 256, 0, stream>>>(
        xc_bf, w_xp_bf, nullptr, nullptr, xdbl_f, nullptr,
        ROWS, DT_RANK + 2 * D_STATE, D_INNER, 0);

    // ---- dt extraction + dt_proj with fused softplus -> delta ----
    extract_dt_kernel<<<(ROWS * DT_RANK) / 256, 256, 0, stream>>>(xdbl_f, dt_bf);
    gemm_wmma_kernel<<<dim3(D_INNER / BN, ROWS / BM), 256, 0, stream>>>(
        dt_bf, w_dt_bf, dt_proj_b, nullptr, delta_f, nullptr,
        ROWS, D_INNER, DT_RANK, 1);

    // ---- selective scan (y overwrites delta) ----
    scan_kernel<<<BB * (D_INNER / 256), 256, 0, stream>>>(xdbl_f, A_log, xc_f, delta_f);

    // ---- gating ----
    gate_kernel<<<(ROWS * D_INNER) / 256, 256, 0, stream>>>(
        delta_f, xc_f, xz_f, D_skip, y_bf);

    // ---- out_proj + residual(x0): x1 = x0 + y * Wop^T ----
    gemm_wmma_kernel<<<dim3(D_MODEL / BN, ROWS / BM), 256, 0, stream>>>(
        y_bf, w_op_bf, nullptr, x0_f, x1_f, nullptr,
        ROWS, D_MODEL, D_INNER, 0);

    // ---- LN2 ----
    ln_single_kernel<<<ROWS, 256, 0, stream>>>(x1_f, ln2_w, ln2_b, h_bf);

    // ---- FFN1 + bias + GELU (bf16 out) ----
    gemm_wmma_kernel<<<dim3(FFN_DIM / BN, ROWS / BM), 256, 0, stream>>>(
        h_bf, w_f1_bf, ffn_b1, nullptr, nullptr, ffnh_bf,
        ROWS, FFN_DIM, D_MODEL, 2);

    // ---- FFN2 + bias + residual(x1) -> d_out ----
    gemm_wmma_kernel<<<dim3(D_MODEL / BN, ROWS / BM), 256, 0, stream>>>(
        ffnh_bf, w_f2_bf, ffn_b2, x1_f, (float*)d_out, nullptr,
        ROWS, D_MODEL, FFN_DIM, 0);
}
